// RenderNet_49048526520747
// MI455X (gfx1250) — matrix operand order repaired and checked by reading, not compiled, not run on hardware
//
#include <hip/hip_runtime.h>

// ---------------------------------------------------------------------------
// RenderNet: line rasterize + 31x31 pyramid conv + global min-max invert.
// Output = 1 - S/Smax (identical across 3 channels), S = conv2d(line, pyramid).
// Cost is dominated by the 201MB output fill -> nontemporal b128 streaming
// stores. Band conv done as im2col-style GEMM on v_wmma_f32_16x16x32_f16.
// ---------------------------------------------------------------------------

typedef _Float16 v16h __attribute__((ext_vector_type(16)));
typedef float    v8f  __attribute__((ext_vector_type(8)));
typedef float    f4   __attribute__((ext_vector_type(4)));

#define IMSIZE 4096
#define PAD    15

__device__ __forceinline__ int imin(int a, int b) { return a < b ? a : b; }
__device__ __forceinline__ int imax(int a, int b) { return a > b ? a : b; }
__device__ __forceinline__ int iabs(int a)        { return a < 0 ? -a : a; }

// ---------------------------------------------------------------------------
// ws layout: [0] Smax (float bits, uint), [64B..] f_lo[4096], f_hi[4096]
// ---------------------------------------------------------------------------
__global__ void init_ws_kernel(int* f_lo, int* f_hi, unsigned* smax_bits) {
    int i = blockIdx.x * blockDim.x + threadIdx.x;
    if (i < IMSIZE) { f_lo[i] = 0x7fffffff; f_hi[i] = -1; }
    if (i == 0) *smax_bits = 0u;  // bits of +0.0f
}

// ---------------------------------------------------------------------------
// Faithful port of the reference Bresenham variant (single thread; ~3800
// sequential steps -> negligible). Records per-row [col_lo, col_hi] so both
// steep and non-steep lines reduce to a per-row column interval.
// ---------------------------------------------------------------------------
__global__ void rasterize_kernel(const float* x0p, const float* y0p,
                                 const float* x1p, const float* y1p,
                                 int* f_lo, int* f_hi) {
    if (threadIdx.x != 0 || blockIdx.x != 0) return;
    float x0 = x0p[0], y0 = y0p[0], x1 = x1p[0], y1 = y1p[0];
    float dx = fabsf(x1 - x0), dy = fabsf(y1 - y0);
    float sx = (x1 > x0) ? 1.f : -1.f;
    float sy = (y1 > y0) ? 1.f : -1.f;
    bool steep = dy > dx;
    float a  = steep ? y0 : x0;
    float b  = steep ? x0 : y0;
    float dX = steep ? dy : dx;
    float dY = steep ? dx : dy;
    float sa = steep ? sy : sx;
    float sb = steep ? sx : sy;
    float d  = 2.f * dY - dX;
    int   n  = (int)dX;
    for (int i = 0; i < n; ++i) {
        int row = (int)(steep ? b : a);
        int col = (int)(steep ? a : b);
        if (row >= 0 && row < IMSIZE && col >= 0 && col < IMSIZE) {
            f_lo[row] = imin(f_lo[row], col);
            f_hi[row] = imax(f_hi[row], col);
        }
        while (d >= 0.f) { b += sb; d -= 2.f * dX; }
        a += sa; d += 2.f * dY;
    }
    int r1 = (int)x1, c1 = (int)y1;
    if (r1 >= 0 && r1 < IMSIZE && c1 >= 0 && c1 < IMSIZE) {
        f_lo[r1] = imin(f_lo[r1], c1);
        f_hi[r1] = imax(f_hi[r1], c1);
    }
}

// ---------------------------------------------------------------------------
// Streaming fill of 1.0f: 201MB of pure stores (> 192MB L2), so nontemporal
// b128 stores are the right policy on MI455X (23.3 TB/s HBM floor ~8.6us).
// Grid-stride: ~12 NT b128 stores per thread keeps store queues deep.
// ---------------------------------------------------------------------------
__global__ void fill_ones_kernel(f4* __restrict__ out, long n4) {
    long idx    = (long)blockIdx.x * blockDim.x + threadIdx.x;
    long stride = (long)gridDim.x * blockDim.x;
    f4 v = {1.f, 1.f, 1.f, 1.f};
    for (long i = idx; i < n4; i += stride)
        __builtin_nontemporal_store(v, out + i);
}

// ---------------------------------------------------------------------------
// Band conv as GEMM on WMMA. One wave (32 threads) per 16-row strip.
// out[r, c0+n] = sum_i sum_u A_i[r,u] * B_i[u,n]
//   A_i[r,u] = [c0-15+u in [f_lo, f_hi] of row r+i-15]   (0/1, exact in f16)
//   B_i[u,n] = max(0, 16 - max(|i-15|, |u-n-15|))/16     (branchless: the
//              clamp-to-0 is exactly the out-of-window condition u-n>30 etc.)
// u padded to 64 -> two v_wmma_f32_16x16x32_f16 per kernel row i.
// B is tile-independent: precomputed once per block into LDS (62 fragments).
// Register packing used by this kernel (A and B share one contiguous k order):
//   A (16x32 f16): lane L -> M = L&15; element e -> K = ((L>>4)<<4) + e
//   B (32x16 f16): lane L -> N = L&15; element e -> K = ((L>>4)<<4) + e
//   C/D (16x16 f32): lane L, vgpr v -> M = v + ((L>>4)<<3), N = L&15
// mode 0: atomicMax of S into smax_bits.  mode 1: write 1 - S/Smax (3 chans).
// ---------------------------------------------------------------------------
__global__ __launch_bounds__(32)
void band_conv_kernel(const int* __restrict__ f_lo, const int* __restrict__ f_hi,
                      unsigned* smax_bits, float* __restrict__ out, int mode) {
    __shared__ v16h sB[62][32];           // 62KB: B fragments for all (i,uoff)
    __shared__ int s_lo[48], s_hi[48];
    __shared__ int s_fmin, s_fmax;
    const int lane = threadIdx.x;         // 0..31 (wave32)
    const int r0 = 16 * blockIdx.x;       // 256 strips cover rows 0..4095
    const int m = lane & 15;              // M for A rows / N for B cols
    const int hh = lane >> 4;             // half-wave selector

    // Stage per-row line intervals for rows r0-15 .. r0+30 into LDS.
    for (int t = lane; t < 46; t += 32) {
        int rr = r0 - 15 + t;
        int lo = 0x7fffffff, hi = -1;
        if (rr >= 0 && rr < IMSIZE) { lo = f_lo[rr]; hi = f_hi[rr]; }
        s_lo[t] = lo; s_hi[t] = hi;
    }
    __syncthreads();
    if (lane == 0) {
        int fmn = 0x7fffffff, fmx = -1;
        for (int t = 0; t < 46; ++t)
            if (s_hi[t] >= s_lo[t]) { fmn = imin(fmn, s_lo[t]); fmx = imax(fmx, s_hi[t]); }
        s_fmin = fmn; s_fmax = fmx;
    }
    __syncthreads();
    const int fmn = s_fmin, fmx = s_fmax;
    if (fmx < 0) return;                  // strip never touches the line band

    // Precompute branchless B fragments (tile-independent) into LDS.
    for (int iu = 0; iu < 62; ++iu) {
        const int i  = iu >> 1;
        const int uo = (iu & 1) << 5;
        v16h b;
        #pragma unroll
        for (int e = 0; e < 16; ++e) {
            int j   = uo + (hh << 4) + e - m;           // kernel col u-n
            int rch = imax(iabs(i - PAD), iabs(j - PAD));
            int v   = imax(0, 16 - rch);                // 0 outside window
            b[e] = (_Float16)((float)v * 0.0625f);
        }
        sB[iu][lane] = b;
    }
    __syncthreads();

    const int c_lo = imax(0, fmn - PAD) & ~15;
    const int c_hi = imin(IMSIZE - 1, fmx + PAD);

    float invS = 0.f;
    if (mode == 1) invS = 1.0f / __uint_as_float(*smax_bits);

    for (int c0 = c_lo; c0 <= c_hi; c0 += 16) {
        v8f acc;
        #pragma unroll
        for (int v = 0; v < 8; ++v) acc[v] = 0.f;

        for (int i = 0; i < 31; ++i) {
            const int lo = s_lo[m + i];   // row r0 + m + i - 15
            const int hi = s_hi[m + i];
            #pragma unroll
            for (int uo = 0; uo < 2; ++uo) {
                v16h a;
                #pragma unroll
                for (int e = 0; e < 16; ++e) {
                    // A: indicator that column (c0-15+u) lies on the line row
                    int k = (hh << 4) + e;
                    int s = c0 - PAD + (uo << 5) + k;
                    a[e] = (s >= lo && s <= hi) ? (_Float16)1.0f : (_Float16)0.0f;
                }
                v16h b = sB[(i << 1) + uo][lane];
                acc = __builtin_amdgcn_wmma_f32_16x16x32_f16(
                    false, a, false, b, (short)0, acc, false, false);
            }
        }

        if (mode == 0) {
            float mx = 0.f;
            #pragma unroll
            for (int v = 0; v < 8; ++v) mx = fmaxf(mx, acc[v]);
            atomicMax(smax_bits, __float_as_uint(mx));   // valid: S >= 0
        } else {
            #pragma unroll
            for (int v = 0; v < 8; ++v) {
                int r = r0 + v + (hh << 3);
                int c = c0 + m;
                if (r < IMSIZE && c < IMSIZE) {
                    float val = 1.0f - acc[v] * invS;
                    size_t p = (size_t)r * IMSIZE + c;
                    out[p]                               = val;
                    out[p + (size_t)IMSIZE * IMSIZE]     = val;
                    out[p + 2 * (size_t)IMSIZE * IMSIZE] = val;
                }
            }
        }
    }
}

// ---------------------------------------------------------------------------
extern "C" void kernel_launch(void* const* d_in, const int* in_sizes, int n_in,
                              void* d_out, int out_size, void* d_ws, size_t ws_size,
                              hipStream_t stream) {
    const float* x0 = (const float*)d_in[0];
    const float* y0 = (const float*)d_in[1];
    const float* x1 = (const float*)d_in[2];
    const float* y1 = (const float*)d_in[3];
    float* out = (float*)d_out;

    unsigned* smax_bits = (unsigned*)d_ws;
    int* f_lo = (int*)((char*)d_ws + 64);
    int* f_hi = f_lo + IMSIZE;

    // 1) reset ws (poisoned once; must re-init every call for determinism)
    init_ws_kernel<<<IMSIZE / 256, 256, 0, stream>>>(f_lo, f_hi, smax_bits);

    // 2) rasterize the line (single wave, sequential Bresenham)
    rasterize_kernel<<<1, 32, 0, stream>>>(x0, y0, x1, y1, f_lo, f_hi);

    // 3) WMMA band conv pass 1: global max of S
    band_conv_kernel<<<IMSIZE / 16, 32, 0, stream>>>(f_lo, f_hi, smax_bits, nullptr, 0);

    // 4) streaming fill of 1.0f (far field: 1 - 0/Smax == 1)
    long n4 = (long)out_size / 4;  // 3*4096*4096 divisible by 4
    fill_ones_kernel<<<4096, 256, 0, stream>>>((f4*)out, n4);

    // 5) WMMA band conv pass 2: overwrite band with 1 - S/Smax (3 channels)
    band_conv_kernel<<<IMSIZE / 16, 32, 0, stream>>>(f_lo, f_hi, smax_bits, out, 1);
}
